// Decoder_29205777613717
// MI455X (gfx1250) — matrix operand order repaired and checked by reading, not compiled, not run on hardware
//
#include <hip/hip_runtime.h>
#include <hip/hip_bf16.h>

// ---------------------------------------------------------------------------
// MI455X (gfx1250) R-GCN decoder.
// Relation messages as DENSE bf16 WMMA GEMMs (Hr = h @ Wr[r] for all r,
// concatenated), edge stage = gather + fp32 atomic scatter-add.
// GEMM staging uses GLOBAL_LOAD_ASYNC_TO_LDS_B128 (memory -> LDS direct,
// ASYNCcnt-pipelined, double buffered); epilogue stages C through LDS for
// coalesced b128 stores.
// ---------------------------------------------------------------------------

typedef __bf16 bf16_t;
typedef __attribute__((ext_vector_type(16))) __bf16 v16bf;
typedef __attribute__((ext_vector_type(8)))  float  v8f;

#define BM 128
#define BN 64
#define BK 32

__device__ __forceinline__ void async_copy_b128(unsigned int lds_addr,
                                                const void* gaddr) {
  asm volatile("global_load_async_to_lds_b128 %0, %1, off"
               :
               : "v"(lds_addr), "v"(gaddr)
               : "memory");
}

// ---------------------------------------------------------------------------
// Tiled GEMM: C[M,N] = A[M,K](bf16 row-major) * Bt[N,K](bf16, B transposed),
// fp32 accum via v_wmma_f32_16x16x32_bf16.
// Block: 256 threads = 8 waves; wave grid 4(M) x 2(N); wave tile 32x32.
// Async double-buffered staging: per thread 3 async b128 per K-step
// (A: 2, B: 1); s_wait_asynccnt 3 overlaps group k+1 with compute of k.
// Requires: K % 32 == 0, N % 64 == 0. M clamped on load, guarded on store.
// ---------------------------------------------------------------------------
template <bool STORE_BF16>
__global__ __launch_bounds__(256) void gemm_bf16_wmma(
    const bf16_t* __restrict__ A, const bf16_t* __restrict__ Bt,
    void* __restrict__ Cout, int Mrows, int K, int Ncols) {
  struct Stage {
    bf16_t As[BM][BK + 8];   // stride 40 bf16 = 80B (16B aligned)
    bf16_t BsT[BN][BK + 8];  // transposed B tile [n][k]
  };
  union Smem {
    Stage  s[2];             // double buffer: 2 x 15360B
    float  cf[BM][BN + 4];   // fp32 C staging (34816B, union max)
    bf16_t cb[BM][BN + 8];   // bf16 C staging
  };
  __shared__ Smem smem;

  const int tid  = threadIdx.x;
  const int lane = tid & 31;
  const int w    = tid >> 5;       // wave 0..7
  const int waveM = w & 3;         // 0..3
  const int waveN = w >> 2;        // 0..1
  const int m0 = blockIdx.y * BM;
  const int n0 = blockIdx.x * BN;

  v8f acc[2][2] = {};

  // cooperative tile-load assignments
  const int arow  = tid >> 1;          // 0..127
  const int ahalf = (tid & 1) * 16;    // 0 or 16
  const int bnrow = tid >> 2;          // 0..63
  const int bpart = (tid & 3) * 8;     // 0,8,16,24

  // branch-free A row clamp: rows beyond M load valid garbage, never stored
  int gr = m0 + arow;
  if (gr > Mrows - 1) gr = Mrows - 1;

  const bf16_t* aG = A + (size_t)gr * K + ahalf;
  const bf16_t* bG = Bt + (size_t)(n0 + bnrow) * K + bpart;
  // LDS byte addresses (flat LDS address truncates to LDS offset)
  const unsigned int aL0 =
      (unsigned int)(uintptr_t)&smem.s[0].As[arow][ahalf];
  const unsigned int bL0 =
      (unsigned int)(uintptr_t)&smem.s[0].BsT[bnrow][bpart];
  const unsigned int bufStride = (unsigned int)sizeof(Stage);

  const int nk = K >> 5;  // K / 32

  // prologue: issue group 0 into buffer 0
  async_copy_b128(aL0, aG);
  async_copy_b128(aL0 + 16, aG + 8);
  async_copy_b128(bL0, bG);

  for (int kk = 0; kk < nk; ++kk) {
    if (kk + 1 < nk) {
      // issue group kk+1 into the other buffer, then wait for group kk
      const unsigned int bo = ((kk + 1) & 1) ? bufStride : 0u;
      const int ko = (kk + 1) << 5;
      async_copy_b128(aL0 + bo, aG + ko);
      async_copy_b128(aL0 + bo + 16, aG + ko + 8);
      async_copy_b128(bL0 + bo, bG + ko);
      asm volatile("s_wait_asynccnt 0x3" ::: "memory");
    } else {
      asm volatile("s_wait_asynccnt 0x0" ::: "memory");
    }
    __syncthreads();  // all waves' group kk now in LDS

    const Stage& st = smem.s[kk & 1];

    // --- fragments per ISA 7.12.2 (wave32) ---
    const int mrow = lane & 15;
    const int ks   = (lane >> 4) * 8;
    v16bf afr[2];
#pragma unroll
    for (int i = 0; i < 2; ++i) {
      union { uint4 u[2]; v16bf v; } fa;
      const int r = waveM * 32 + i * 16 + mrow;
      fa.u[0] = *reinterpret_cast<const uint4*>(&st.As[r][ks]);
      fa.u[1] = *reinterpret_cast<const uint4*>(&st.As[r][ks + 16]);
      afr[i] = fa.v;
    }
    const int ncol = lane & 15;
    const int kb   = (lane >> 4) * 8;
    v16bf bfr[2];
#pragma unroll
    for (int j = 0; j < 2; ++j) {
      union { uint4 u[2]; v16bf v; } fb;
      const int c = waveN * 32 + j * 16 + ncol;
      fb.u[0] = *reinterpret_cast<const uint4*>(&st.BsT[c][kb]);
      fb.u[1] = *reinterpret_cast<const uint4*>(&st.BsT[c][kb + 16]);
      bfr[j] = fb.v;
    }

#pragma unroll
    for (int i = 0; i < 2; ++i)
#pragma unroll
      for (int j = 0; j < 2; ++j)
        acc[i][j] = __builtin_amdgcn_wmma_f32_16x16x32_bf16(
            false, afr[i], false, bfr[j], (short)0, acc[i][j], false, false);
    __syncthreads();  // done reading buffer kk before it is overwritten
  }

  // --- epilogue: stage C tile in LDS, then coalesced b128 stores ---
  // C/D frag layout: row = g + 8*(lane>=16), col = lane%16 (per 16x16 frag)
  const int rloc0 = waveM * 32 + ((lane >> 4) << 3);
  const int cloc0 = waveN * 32 + (lane & 15);
  if constexpr (!STORE_BF16) {
#pragma unroll
    for (int i = 0; i < 2; ++i)
#pragma unroll
      for (int j = 0; j < 2; ++j)
#pragma unroll
        for (int g = 0; g < 8; ++g)
          smem.cf[rloc0 + i * 16 + g][cloc0 + j * 16] = acc[i][j][g];
    __syncthreads();
    float* Cf = reinterpret_cast<float*>(Cout);
#pragma unroll
    for (int it = 0; it < 8; ++it) {
      const int linear = tid + it * 256;    // 128 rows x 16 float4
      const int row = linear >> 4;
      const int q   = (linear & 15) * 4;
      if (m0 + row < Mrows) {
        const float4 v = *reinterpret_cast<const float4*>(&smem.cf[row][q]);
        *reinterpret_cast<float4*>(&Cf[(size_t)(m0 + row) * Ncols + n0 + q]) = v;
      }
    }
  } else {
#pragma unroll
    for (int i = 0; i < 2; ++i)
#pragma unroll
      for (int j = 0; j < 2; ++j)
#pragma unroll
        for (int g = 0; g < 8; ++g)
          smem.cb[rloc0 + i * 16 + g][cloc0 + j * 16] = (bf16_t)acc[i][j][g];
    __syncthreads();
    bf16_t* Cb = reinterpret_cast<bf16_t*>(Cout);
#pragma unroll
    for (int it = 0; it < 4; ++it) {
      const int linear = tid + it * 256;    // 128 rows x 8 uint4 (8 bf16 each)
      const int row = linear >> 3;
      const int q   = (linear & 7) * 8;
      if (m0 + row < Mrows) {
        const uint4 v = *reinterpret_cast<const uint4*>(&smem.cb[row][q]);
        *reinterpret_cast<uint4*>(&Cb[(size_t)(m0 + row) * Ncols + n0 + q]) = v;
      }
    }
  }
}

// ---------------------------------------------------------------------------
// Per-graph context mean: x_batch sorted -> run-length accumulate in
// registers, one atomic per (graph,column) run boundary. Thread = column.
// ---------------------------------------------------------------------------
__global__ void ctx_accum_kernel(const float* __restrict__ x,
                                 const int* __restrict__ xb,
                                 float* __restrict__ ctx_sum,
                                 float* __restrict__ ctx_cnt, int Mrows,
                                 int rowsPerBlock) {
  const int t  = threadIdx.x;
  const int r0 = blockIdx.x * rowsPerBlock;
  int r1 = r0 + rowsPerBlock;
  if (r1 > Mrows) r1 = Mrows;
  int cur = -1;
  float s = 0.f, c = 0.f;
  for (int r = r0; r < r1; ++r) {
    const int g = xb[r];
    if (g != cur) {
      if (cur >= 0) {
        atomicAdd(&ctx_sum[cur * 256 + t], s);
        if (t == 0) atomicAdd(&ctx_cnt[cur], c);
      }
      cur = g; s = 0.f; c = 0.f;
    }
    s += x[(size_t)r * 256 + t];
    c += 1.f;
  }
  if (cur >= 0) {
    atomicAdd(&ctx_sum[cur * 256 + t], s);
    if (t == 0) atomicAdd(&ctx_cnt[cur], c);
  }
}

__global__ void ctx_final_kernel(float* __restrict__ ctx,
                                 const float* __restrict__ cnt) {
  const int i = blockIdx.x * blockDim.x + threadIdx.x;  // 64*256
  const float c = cnt[i >> 8];
  ctx[i] = ctx[i] / fmaxf(c, 1.f);
}

// ctxW[g, j] = b[j] + sum_k ctx[g,k] * Wc[k,j]   (tiny: 64 x dout)
__global__ void ctxw_kernel(const float* __restrict__ ctx,
                            const float* __restrict__ Wc,
                            const float* __restrict__ bias,
                            float* __restrict__ ctxW, int dout) {
  const int i = blockIdx.x * blockDim.x + threadIdx.x;
  if (i >= 64 * dout) return;
  const int g = i / dout, j = i - g * dout;
  float s = bias[j];
  const float* cr = ctx + g * 256;
  for (int k = 0; k < 256; ++k) s += cr[k] * Wc[(size_t)k * dout + j];
  ctxW[i] = s;
}

// W [K, Nc] fp32 -> Wt [Nc, K] bf16 (transposed for WMMA B path)
__global__ void conv_bf16_T_kernel(const float* __restrict__ src,
                                   bf16_t* __restrict__ dst, int K, int Nc) {
  const size_t total = (size_t)K * Nc;
  const size_t i = blockIdx.x * (size_t)blockDim.x + threadIdx.x;
  if (i >= total) return;
  const int k = (int)(i / Nc);
  const int n = (int)(i - (size_t)k * Nc);
  dst[(size_t)n * K + k] = (bf16_t)src[i];
}

// Wr [4, din, dout] fp32 -> WcatT [4*dout, din] bf16
__global__ void conv_relcat_T_kernel(const float* __restrict__ Wr,
                                     bf16_t* __restrict__ dst, int din,
                                     int dout) {
  const size_t total = (size_t)4 * din * dout;
  const size_t i = blockIdx.x * (size_t)blockDim.x + threadIdx.x;
  if (i >= total) return;
  const int r = (int)(i / ((size_t)din * dout));
  const size_t rem = i - (size_t)r * din * dout;
  const int k = (int)(rem / dout);
  const int j = (int)(rem - (size_t)k * dout);
  dst[((size_t)r * dout + j) * din + k] = (bf16_t)Wr[i];
}

// h0[n,j] = sum_{k<3} embeds[tgt_y[n,k], j]   -> bf16
__global__ void embed_init_kernel(const int* __restrict__ tgt_y,
                                  const float* __restrict__ embeds,
                                  bf16_t* __restrict__ h0, int Nn) {
  const size_t i = blockIdx.x * (size_t)blockDim.x + threadIdx.x;
  if (i >= (size_t)Nn * 128) return;
  const int n = (int)(i >> 7), j = (int)(i & 127);
  const int* t = tgt_y + (size_t)n * 3;
  const float s = embeds[(size_t)t[0] * 128 + j] +
                  embeds[(size_t)t[1] * 128 + j] +
                  embeds[(size_t)t[2] * 128 + j];
  h0[i] = (bf16_t)s;
}

__global__ void add_ctx_kernel(float* __restrict__ pre,
                               const float* __restrict__ ctxW,
                               const int* __restrict__ yb, int Nn, int dout) {
  const size_t i = blockIdx.x * (size_t)blockDim.x + threadIdx.x;
  if (i >= (size_t)Nn * dout) return;
  const int n = (int)(i / dout);
  const int j = (int)(i - (size_t)n * dout);
  pre[i] += ctxW[(size_t)yb[n] * dout + j];
}

// pre[dst] += Hrel[src, etype-block]; 4 columns per thread
__global__ void edge_scatter_kernel(const bf16_t* __restrict__ Hrel,
                                    float* __restrict__ pre,
                                    const int* __restrict__ src,
                                    const int* __restrict__ dst,
                                    const int* __restrict__ et, int Ee,
                                    int dout) {
  const int qper = dout >> 2;
  const size_t i = blockIdx.x * (size_t)blockDim.x + threadIdx.x;
  if (i >= (size_t)Ee * qper) return;
  const int e = (int)(i / qper);
  const int q = ((int)(i - (size_t)e * qper)) << 2;
  const int s = src[e], d = dst[e], r = et[e];
  const bf16_t* hp = Hrel + ((size_t)s * 4 + r) * dout + q;
  float* pp = pre + (size_t)d * dout + q;
  atomicAdd(pp + 0, (float)hp[0]);
  atomicAdd(pp + 1, (float)hp[1]);
  atomicAdd(pp + 2, (float)hp[2]);
  atomicAdd(pp + 3, (float)hp[3]);
}

__global__ void relu_kernel(const float* __restrict__ pre,
                            bf16_t* __restrict__ hnext,
                            float* __restrict__ yout, size_t n) {
  const size_t i = blockIdx.x * (size_t)blockDim.x + threadIdx.x;
  if (i >= n) return;
  const float v = fmaxf(pre[i], 0.f);
  hnext[i] = (bf16_t)v;
  if (yout) yout[i] = v;
}

// in-place log_softmax over 512 cols (one wave32 per row, 16 elems/lane)
__global__ void logsoftmax512_kernel(float* __restrict__ logits,
                                     const float* __restrict__ bias, int Nn) {
  const int wid  = (int)((blockIdx.x * (size_t)blockDim.x + threadIdx.x) >> 5);
  const int lane = threadIdx.x & 31;
  if (wid >= Nn) return;
  float* row = logits + (size_t)wid * 512;
  float v[16];
  float m = -3.0e38f;
#pragma unroll
  for (int i = 0; i < 16; ++i) {
    v[i] = row[lane + i * 32] + bias[lane + i * 32];
    m = fmaxf(m, v[i]);
  }
#pragma unroll
  for (int o = 16; o > 0; o >>= 1) m = fmaxf(m, __shfl_xor(m, o, 32));
  float s = 0.f;
#pragma unroll
  for (int i = 0; i < 16; ++i) s += __expf(v[i] - m);
#pragma unroll
  for (int o = 16; o > 0; o >>= 1) s += __shfl_xor(s, o, 32);
  const float ls = __logf(s);
#pragma unroll
  for (int i = 0; i < 16; ++i) row[lane + i * 32] = v[i] - m - ls;
}

// edge head: log_softmax(concat(y[src],y[dst]) @ Wg + bg) -> [E,5]
__global__ void edge_head_kernel(const float* __restrict__ y,
                                 const int* __restrict__ src,
                                 const int* __restrict__ dst,
                                 const float* __restrict__ Wg,
                                 const float* __restrict__ bg,
                                 float* __restrict__ out, int Ee) {
  const int wid  = (int)((blockIdx.x * (size_t)blockDim.x + threadIdx.x) >> 5);
  const int lane = threadIdx.x & 31;
  if (wid >= Ee) return;
  const float* ys = y + (size_t)src[wid] * 128;
  const float* yd = y + (size_t)dst[wid] * 128;
  float acc[5] = {0.f, 0.f, 0.f, 0.f, 0.f};
  for (int k = lane; k < 128; k += 32) {
    const float a = ys[k], b = yd[k];
    const float* w0 = Wg + (size_t)k * 5;
    const float* w1 = Wg + (size_t)(128 + k) * 5;
#pragma unroll
    for (int c = 0; c < 5; ++c) acc[c] += a * w0[c] + b * w1[c];
  }
#pragma unroll
  for (int c = 0; c < 5; ++c)
#pragma unroll
    for (int o = 16; o > 0; o >>= 1) acc[c] += __shfl_xor(acc[c], o, 32);
  if (lane == 0) {
    float t[5], m = -3.0e38f;
#pragma unroll
    for (int c = 0; c < 5; ++c) {
      t[c] = acc[c] + bg[c];
      m = fmaxf(m, t[c]);
    }
    float s = 0.f;
#pragma unroll
    for (int c = 0; c < 5; ++c) s += __expf(t[c] - m);
    const float ls = __logf(s);
    float* o5 = out + (size_t)wid * 5;
#pragma unroll
    for (int c = 0; c < 5; ++c) o5[c] = t[c] - m - ls;
  }
}

// ---------------------------------------------------------------------------
static inline size_t cdiv(size_t a, size_t b) { return (a + b - 1) / b; }

extern "C" void kernel_launch(void* const* d_in, const int* in_sizes, int n_in,
                              void* d_out, int out_size, void* d_ws,
                              size_t ws_size, hipStream_t stream) {
  const int M = in_sizes[0] / 256;   // encoder nodes
  const int N = in_sizes[2] / 3;     // decoder nodes
  const int E = in_sizes[4];         // edges

  const float* x       = (const float*)d_in[0];
  const int*   x_batch = (const int*)d_in[1];
  const int*   tgt_y   = (const int*)d_in[2];
  const int*   e_index = (const int*)d_in[3];
  const int*   e_type  = (const int*)d_in[4];
  const int*   y_batch = (const int*)d_in[5];
  const float* embeds  = (const float*)d_in[6];
  const float* Ws1 = (const float*)d_in[7];
  const float* Wr1 = (const float*)d_in[8];
  const float* Wc1 = (const float*)d_in[9];
  const float* b1  = (const float*)d_in[10];
  const float* Ws2 = (const float*)d_in[11];
  const float* Wr2 = (const float*)d_in[12];
  const float* Wc2 = (const float*)d_in[13];
  const float* b2  = (const float*)d_in[14];
  const float* Ws3 = (const float*)d_in[15];
  const float* Wr3 = (const float*)d_in[16];
  const float* Wc3 = (const float*)d_in[17];
  const float* b3  = (const float*)d_in[18];
  const float* Wz  = (const float*)d_in[19];
  const float* bz  = (const float*)d_in[20];
  const float* Wg  = (const float*)d_in[21];
  const float* bg  = (const float*)d_in[22];

  const int* src = e_index;       // e_index[0,:]
  const int* dst = e_index + E;   // e_index[1,:]

  // output layout (flat, return order)
  float* out_y  = (float*)d_out;                              // [N,128]
  char*  out_ei = (char*)d_out + (size_t)N * 128 * 4;         // [2,E] int
  char*  out_et = out_ei + (size_t)2 * E * 4;                 // [E] int
  float* out_yp = (float*)(out_et + (size_t)E * 4);           // [N,512]
  float* out_ep = out_yp + (size_t)N * 512;                   // [E,5]

  // workspace carve-out (256B aligned)
  char* ws = (char*)d_ws;
  size_t off = 0;
  auto alloc = [&](size_t bytes) -> void* {
    void* p = ws + off;
    off = (off + bytes + 255) & ~(size_t)255;
    return p;
  };
  float*  ctx   = (float*)alloc(64 * 256 * 4);
  float*  cnt   = (float*)alloc(64 * 4);
  float*  ctxW1 = (float*)alloc(64 * 256 * 4);
  float*  ctxW2 = (float*)alloc(64 * 256 * 4);
  float*  ctxW3 = (float*)alloc(64 * 128 * 4);
  bf16_t* Ws1b  = (bf16_t*)alloc((size_t)128 * 256 * 2);   // [256,128] T
  bf16_t* Ws2b  = (bf16_t*)alloc((size_t)256 * 256 * 2);   // [256,256] T
  bf16_t* Ws3b  = (bf16_t*)alloc((size_t)256 * 128 * 2);   // [128,256] T
  bf16_t* Wr1b  = (bf16_t*)alloc((size_t)128 * 1024 * 2);  // [1024,128] T
  bf16_t* Wr2b  = (bf16_t*)alloc((size_t)256 * 1024 * 2);  // [1024,256] T
  bf16_t* Wr3b  = (bf16_t*)alloc((size_t)256 * 512 * 2);   // [512,256] T
  bf16_t* Wzb   = (bf16_t*)alloc((size_t)128 * 512 * 2);   // [512,128] T
  bf16_t* hA    = (bf16_t*)alloc((size_t)N * 256 * 2);
  bf16_t* hB    = (bf16_t*)alloc((size_t)N * 256 * 2);
  float*  pre   = (float*)alloc((size_t)N * 256 * 4);
  bf16_t* Hrel  = (bf16_t*)alloc((size_t)N * 1024 * 2);
  (void)ws_size;

  const int T = 256;

  // ---- context mean + per-layer ctx projections (tiny) ----
  hipMemsetAsync(ctx, 0, 64 * 256 * 4, stream);
  hipMemsetAsync(cnt, 0, 64 * 4, stream);
  {
    const int rpb = 128;
    ctx_accum_kernel<<<(int)cdiv(M, rpb), 256, 0, stream>>>(x, x_batch, ctx,
                                                            cnt, M, rpb);
    ctx_final_kernel<<<64, 256, 0, stream>>>(ctx, cnt);
    ctxw_kernel<<<(int)cdiv(64 * 256, T), T, 0, stream>>>(ctx, Wc1, b1, ctxW1, 256);
    ctxw_kernel<<<(int)cdiv(64 * 256, T), T, 0, stream>>>(ctx, Wc2, b2, ctxW2, 256);
    ctxw_kernel<<<(int)cdiv(64 * 128, T), T, 0, stream>>>(ctx, Wc3, b3, ctxW3, 128);
  }

  // ---- weight conversion to transposed bf16 (cheap, once per call) ----
  conv_bf16_T_kernel<<<(int)cdiv(128 * 256, T), T, 0, stream>>>(Ws1, Ws1b, 128, 256);
  conv_bf16_T_kernel<<<(int)cdiv(256 * 256, T), T, 0, stream>>>(Ws2, Ws2b, 256, 256);
  conv_bf16_T_kernel<<<(int)cdiv(256 * 128, T), T, 0, stream>>>(Ws3, Ws3b, 256, 128);
  conv_bf16_T_kernel<<<(int)cdiv(128 * 512, T), T, 0, stream>>>(Wz, Wzb, 128, 512);
  conv_relcat_T_kernel<<<(int)cdiv((size_t)4 * 128 * 256, T), T, 0, stream>>>(Wr1, Wr1b, 128, 256);
  conv_relcat_T_kernel<<<(int)cdiv((size_t)4 * 256 * 256, T), T, 0, stream>>>(Wr2, Wr2b, 256, 256);
  conv_relcat_T_kernel<<<(int)cdiv((size_t)4 * 256 * 128, T), T, 0, stream>>>(Wr3, Wr3b, 256, 128);

  // ---- h0 = sum of 3 embedding rows ----
  embed_init_kernel<<<(int)cdiv((size_t)N * 128, T), T, 0, stream>>>(tgt_y, embeds, hA, N);

  const int gy = (int)cdiv(N, BM);  // GEMM M-tiles

  // Layer 1: din=128, dout=256, in=hA, out=hB
  {
    dim3 gr(1024 / BN, gy);
    gemm_bf16_wmma<true><<<gr, 256, 0, stream>>>(hA, Wr1b, Hrel, N, 128, 1024);
    dim3 gs(256 / BN, gy);
    gemm_bf16_wmma<false><<<gs, 256, 0, stream>>>(hA, Ws1b, pre, N, 128, 256);
    add_ctx_kernel<<<(int)cdiv((size_t)N * 256, T), T, 0, stream>>>(pre, ctxW1, y_batch, N, 256);
    edge_scatter_kernel<<<(int)cdiv((size_t)E * 64, T), T, 0, stream>>>(Hrel, pre, src, dst, e_type, E, 256);
    relu_kernel<<<(int)cdiv((size_t)N * 256, T), T, 0, stream>>>(pre, hB, nullptr, (size_t)N * 256);
  }
  // Layer 2: din=256, dout=256, in=hB, out=hA
  {
    dim3 gr(1024 / BN, gy);
    gemm_bf16_wmma<true><<<gr, 256, 0, stream>>>(hB, Wr2b, Hrel, N, 256, 1024);
    dim3 gs(256 / BN, gy);
    gemm_bf16_wmma<false><<<gs, 256, 0, stream>>>(hB, Ws2b, pre, N, 256, 256);
    add_ctx_kernel<<<(int)cdiv((size_t)N * 256, T), T, 0, stream>>>(pre, ctxW2, y_batch, N, 256);
    edge_scatter_kernel<<<(int)cdiv((size_t)E * 64, T), T, 0, stream>>>(Hrel, pre, src, dst, e_type, E, 256);
    relu_kernel<<<(int)cdiv((size_t)N * 256, T), T, 0, stream>>>(pre, hA, nullptr, (size_t)N * 256);
  }
  // Layer 3: din=256, dout=128, in=hA, out=hB (+ fp32 y to d_out)
  {
    dim3 gr(512 / BN, gy);
    gemm_bf16_wmma<true><<<gr, 256, 0, stream>>>(hA, Wr3b, Hrel, N, 256, 512);
    dim3 gs(128 / BN, gy);
    gemm_bf16_wmma<false><<<gs, 256, 0, stream>>>(hA, Ws3b, pre, N, 256, 128);
    add_ctx_kernel<<<(int)cdiv((size_t)N * 128, T), T, 0, stream>>>(pre, ctxW3, y_batch, N, 128);
    edge_scatter_kernel<<<(int)cdiv((size_t)E * 32, T), T, 0, stream>>>(Hrel, pre, src, dst, e_type, E, 128);
    relu_kernel<<<(int)cdiv((size_t)N * 128, T), T, 0, stream>>>(pre, hB, out_y, (size_t)N * 128);
  }

  // ---- node head: logits straight into d_out, then in-place log_softmax ----
  {
    dim3 gz(512 / BN, gy);
    gemm_bf16_wmma<false><<<gz, 256, 0, stream>>>(hB, Wzb, out_yp, N, 128, 512);
    logsoftmax512_kernel<<<(int)cdiv((size_t)N * 32, T), T, 0, stream>>>(out_yp, bz, N);
  }

  // ---- edge head ----
  edge_head_kernel<<<(int)cdiv((size_t)E * 32, T), T, 0, stream>>>(out_y, src, dst, Wg, bg, out_ep, E);

  // ---- int passthrough outputs (bitwise copy) ----
  hipMemcpyAsync(out_ei, e_index, (size_t)2 * E * 4, hipMemcpyDeviceToDevice, stream);
  hipMemcpyAsync(out_et, e_type, (size_t)E * 4, hipMemcpyDeviceToDevice, stream);
}
